// PhyloATTN_24867860643969
// MI455X (gfx1250) — compile-verified, matrix operand order checked by reading, not run on hardware
//
#include <hip/hip_runtime.h>
#include <hip/hip_bf16.h>

typedef __bf16 bf16_t;
typedef __attribute__((ext_vector_type(16))) __bf16 v16bf;
typedef __attribute__((ext_vector_type(8)))  float  v8f;

#define BATCH 8
#define RR 64
#define CC 16
#define DD 128
#define NPAIR 2016
#define NP 8            // pairs per block
#define NTILES 252      // NPAIR / NP

// ---------------------------------------------------------------------------
// WMMA bf16 fragment load. A-matrix 16x32 bf16 layout (ISA 7.12.2):
//   lane = half*16 + m ; VGPR0-3 hold K = half*8 .. half*8+7 (pairs),
//   VGPR4-7 hold K = 16 + half*8 .. 16 + half*8 + 7.
// With row-major [row][k] storage each lane reads two contiguous 16B chunks.
// Weights are stored pre-transposed ([n][k]) so B fragments use the same path.
// ---------------------------------------------------------------------------
union Frag16 { v16bf v; uint4 q[2]; };

__device__ __forceinline__ v16bf frag_ld(const bf16_t* rowptr, int kbase) {
  const int half = (threadIdx.x & 31) >> 4;
  const uint4* p = (const uint4*)(rowptr + kbase + half * 8);
  Frag16 f;
  f.q[0] = p[0];
  f.q[1] = p[2];   // +16 bf16 elements = +32B
  return f.v;
}

template<int KSTEPS>
__device__ __forceinline__ v8f mm_tile(const bf16_t* arow, const bf16_t* brow) {
  v8f acc = {};
#pragma unroll
  for (int ks = 0; ks < KSTEPS; ++ks) {
    v16bf a = frag_ld(arow, ks * 32);
    v16bf b = frag_ld(brow, ks * 32);
    acc = __builtin_amdgcn_wmma_f32_16x16x32_bf16(
        /*neg_a=*/false, a, /*neg_b=*/false, b,
        /*c_mod=*/(short)0, acc, /*reuse_a=*/false, /*reuse_b=*/false);
  }
  return acc;
}

// Fast reciprocal (v_rcp_f32, no IEEE div fixup chain) -- ~1 ulp, far below
// the bf16 quantization already accepted in the GEMM inputs.
__device__ __forceinline__ float fast_rcp(float x) {
  return __builtin_amdgcn_rcpf(x);
}
__device__ __forceinline__ float sigmoidf_(float x) {
  return fast_rcp(1.0f + __expf(-x));
}

// ---------------------------------------------------------------------------
// Prep kernels (tiny, all L2-resident)
// ---------------------------------------------------------------------------
__global__ void prep_wt(const float* __restrict__ hW, const float* __restrict__ qW,
                        const float* __restrict__ gW, const float* __restrict__ s1W,
                        bf16_t* __restrict__ hWt, bf16_t* __restrict__ qWt,
                        bf16_t* __restrict__ gWt, bf16_t* __restrict__ s1Wt) {
  int idx = blockIdx.x * blockDim.x + threadIdx.x;   // 4 * 16384
  int which = idx >> 14;
  int e = idx & 16383;                               // e = n*128 + k
  const float* src = which == 0 ? hW : which == 1 ? qW : which == 2 ? gW : s1W;
  bf16_t* dst = which == 0 ? hWt : which == 1 ? qWt : which == 2 ? gWt : s1Wt;
  dst[e] = (bf16_t)src[(e & 127) * 128 + (e >> 7)];  // Wt[n][k] = W[k][n]
}

__global__ void prep_inpT(const float* __restrict__ inp, bf16_t* __restrict__ inpT) {
  int idx = blockIdx.x * blockDim.x + threadIdx.x;   // 8 * 2048 * 64
  int r  = idx & 63;
  int cd = (idx >> 6) & 2047;
  int b  = idx >> 17;
  inpT[idx] = (bf16_t)inp[((size_t)b * RR + r) * 2048 + cd];
}

__global__ void prep_k(const float* __restrict__ inp, const float* __restrict__ kW,
                       const float* __restrict__ kb, bf16_t* __restrict__ kbf) {
  int idx = blockIdx.x * blockDim.x + threadIdx.x;   // 8*64*2048 = 2^20
  int dout = idx & 127;
  const float* x = inp + (size_t)(idx >> 7) * 128;   // row (b, r, c)
  float acc = kb[dout];
#pragma unroll 8
  for (int kk = 0; kk < 128; ++kk) acc += x[kk] * kW[kk * 128 + dout];
  kbf[idx] = (bf16_t)acc;                            // kbf[(b*64+r)*2048 + c*128 + d]
}

// ---------------------------------------------------------------------------
// Fused main kernel: one block per (batch b, tile of 8 pairs).
// Wave w owns pair np0+w (rows [w*16, w*16+16) of the 128x128 activation tile).
// ---------------------------------------------------------------------------
__global__ __launch_bounds__(256)
void phylo_main(const float* __restrict__ inp, const float* __restrict__ seq_mask,
                const int* __restrict__ rowIdx, const int* __restrict__ colIdx,
                const float* __restrict__ hb, const float* __restrict__ gb,
                const float* __restrict__ qb, const float* __restrict__ s1b,
                const float* __restrict__ s2W, const float* __restrict__ s2b,
                const bf16_t* __restrict__ hWt, const bf16_t* __restrict__ qWt,
                const bf16_t* __restrict__ gWt, const bf16_t* __restrict__ s1Wt,
                const bf16_t* __restrict__ kbf, const bf16_t* __restrict__ inpT,
                float* __restrict__ out) {
  __shared__ __align__(16) bf16_t sX[128 * 128];   // 32 KB, multipurpose
  __shared__ __align__(16) bf16_t sY[128 * 128];   // 32 KB, holds x
  __shared__ __align__(16) float  sAf[16 * 64];    // 4 KB, alpha logits / bf16 alpha
  __shared__ int sRow[NP], sCol[NP];

  const int b    = blockIdx.x / NTILES;
  const int np0  = (blockIdx.x % NTILES) * NP;
  const int tid  = threadIdx.x;
  const int wave = tid >> 5;
  const int lane = tid & 31;
  const int lm   = lane & 15;
  const int lh   = lane >> 4;

  if (tid < NP) { sRow[tid] = rowIdx[np0 + tid]; sCol[tid] = colIdx[np0 + tid]; }
  __syncthreads();

  const size_t inp_b = (size_t)b * RR * CC * DD;

  // Stage 1: sX = bf16(x_i - x_j)
  for (int idx = tid; idx < 128 * 128; idx += 256) {
    int rrow = idx >> 7, dcol = idx & 127;
    int p = rrow >> 4, c = rrow & 15;
    float xi = inp[inp_b + (size_t)(sRow[p] * CC + c) * DD + dcol];
    float xj = inp[inp_b + (size_t)(sCol[p] * CC + c) * DD + dcol];
    sX[idx] = (bf16_t)(xi - xj);
  }
  __syncthreads();

  const bf16_t* aX = &sX[(wave * 16 + lm) * 128];
  const bf16_t* aY = &sY[(wave * 16 + lm) * 128];
  bf16_t* wX = &sX[wave * 16 * 128];
  bf16_t* wY = &sY[wave * 16 * 128];
  const float* xi_base = &inp[inp_b + (size_t)sRow[wave] * CC * DD];
  const float* xj_base = &inp[inp_b + (size_t)sCol[wave] * CC * DD];

  // Stage 2: h = diff@hW + hb ; z = sigmoid(h) ; x = z*xi + (1-z)*xj -> sY
  for (int nt = 0; nt < 8; ++nt) {
    v8f acc = mm_tile<4>(aX, &hWt[(nt * 16 + lm) * 128]);
#pragma unroll
    for (int r = 0; r < 8; ++r) {
      int m = lh * 8 + r, d = nt * 16 + lm;
      float z = sigmoidf_(acc[r] + hb[d]);
      float xv = z * xi_base[m * 128 + d] + (1.0f - z) * xj_base[m * 128 + d];
      wY[m * 128 + d] = (bf16_t)xv;
    }
  }
  __syncthreads();

  // Stage 3: q = x@qW + qb -> sX
  for (int nt = 0; nt < 8; ++nt) {
    v8f acc = mm_tile<4>(aY, &qWt[(nt * 16 + lm) * 128]);
#pragma unroll
    for (int r = 0; r < 8; ++r) {
      int m = lh * 8 + r, d = nt * 16 + lm;
      wX[m * 128 + d] = (bf16_t)(acc[r] + qb[d]);
    }
  }
  __syncthreads();

  // Stage 4: alpha[p, r] = scale * sum_{cd} q[p,cd] * k[b,r,cd]
  // A rows = pairs (stride 2048, pairs 8..15 duplicated & ignored), K=2048, N=64.
  if (wave < 4) {
    const bf16_t* aq = &sX[(lm & 7) * 2048];
    const bf16_t* bk = &kbf[((size_t)b * RR + wave * 16 + lm) * 2048];
    v8f acc = mm_tile<64>(aq, bk);
    const float scale = 0.022097086912079612f;   // 1/sqrt(D*C) = 1/sqrt(2048)
#pragma unroll
    for (int r = 0; r < 8; ++r)
      sAf[(lh * 8 + r) * 64 + wave * 16 + lm] = acc[r] * scale;
  }
  __syncthreads();

  // Stage 5: masked softmax over r (64) per pair; wave w handles pair w.
  {
    int r0 = lane, r1 = lane + 32;
    float a0 = sAf[wave * 64 + r0];
    float a1 = sAf[wave * 64 + r1];
    int ri = sRow[wave], ci = sCol[wave];
    const float NEG = -3.0e38f;
    if (r0 == ri || r0 == ci) a0 = NEG;
    if (r1 == ri || r1 == ci) a1 = NEG;
    float mx = fmaxf(a0, a1);
    for (int off = 16; off > 0; off >>= 1) mx = fmaxf(mx, __shfl_xor(mx, off, 32));
    float e0 = __expf(a0 - mx), e1 = __expf(a1 - mx);
    float s = e0 + e1;
    for (int off = 16; off > 0; off >>= 1) s += __shfl_xor(s, off, 32);
    float invs = fast_rcp(s);
    __syncthreads();                       // all reads of sAf done; reuse as bf16
    bf16_t* sAB = (bf16_t*)sAf;
    sAB[wave * 64 + r0] = (bf16_t)(e0 * invs);
    sAB[wave * 64 + r1] = (bf16_t)(e1 * invs);
    sAB[(8 + wave) * 64 + r0] = (bf16_t)0.0f;   // pad pairs 8..15 with zeros
    sAB[(8 + wave) * 64 + r1] = (bf16_t)0.0f;
  }
  __syncthreads();

  // Stage 6: x_glob[p, cd] = sum_r alpha[p,r] * inp[b,r,cd]  (M=16, K=64, N=2048)
  {
    const bf16_t* sAB = (const bf16_t*)sAf;
    const bf16_t* aA = &sAB[lm * 64];
    for (int t = 0; t < 16; ++t) {
      int nt = wave * 16 + t;                 // n-tile 0..127
      v8f acc = mm_tile<2>(aA, &inpT[((size_t)b * 2048 + nt * 16 + lm) * 64]);
#pragma unroll
      for (int r = 0; r < 8; ++r) {
        int p = lh * 8 + r;
        if (p < 8) {
          int cd = nt * 16 + lm;
          sX[(p * 16 + (cd >> 7)) * 128 + (cd & 127)] = (bf16_t)acc[r];
        }
      }
    }
  }
  __syncthreads();

  // Stage 7: g = x_glob@gW + gb ; w = sigmoid(g) ; x = (1-w)*x + w*x_glob -> sY
  for (int nt = 0; nt < 8; ++nt) {
    v8f acc = mm_tile<4>(aX, &gWt[(nt * 16 + lm) * 128]);
#pragma unroll
    for (int r = 0; r < 8; ++r) {
      int m = lh * 8 + r, d = nt * 16 + lm;
      float w8 = sigmoidf_(acc[r] + gb[d]);
      float xg = (float)wX[m * 128 + d];
      float xo = (float)wY[m * 128 + d];
      wY[m * 128 + d] = (bf16_t)((1.0f - w8) * xo + w8 * xg);
    }
  }
  __syncthreads();

  // Stage 8: t = gelu(x@s1W + s1b) (exact erf gelu) -> sX
  for (int nt = 0; nt < 8; ++nt) {
    v8f acc = mm_tile<4>(aY, &s1Wt[(nt * 16 + lm) * 128]);
#pragma unroll
    for (int r = 0; r < 8; ++r) {
      int m = lh * 8 + r, d = nt * 16 + lm;
      float t1 = acc[r] + s1b[d];
      float ge = 0.5f * t1 * (1.0f + erff(t1 * 0.7071067811865475f));
      wX[m * 128 + d] = (bf16_t)ge;
    }
  }
  __syncthreads();

  // Stage 9: s = t@s2W + s2b ; out[b, n] = sum_c s[c] * seq_mask[b, c]
  {
    float part = 0.0f;
    const bf16_t* rowp = &sX[(wave * 16 + lm) * 128 + lh * 64];
    const float* s2p = s2W + lh * 64;
#pragma unroll 16
    for (int d = 0; d < 64; ++d) part += (float)rowp[d] * s2p[d];
    part += __shfl_xor(part, 16, 32);                 // combine the two d-halves
    float sc = (part + s2b[0]) * seq_mask[b * CC + lm];
    for (int off = 8; off > 0; off >>= 1) sc += __shfl_xor(sc, off, 32);
    if (lane == 0) out[(size_t)b * NPAIR + np0 + wave] = sc;
  }
}

// ---------------------------------------------------------------------------
extern "C" void kernel_launch(void* const* d_in, const int* in_sizes, int n_in,
                              void* d_out, int out_size, void* d_ws, size_t ws_size,
                              hipStream_t stream) {
  (void)in_sizes; (void)n_in; (void)out_size; (void)ws_size;
  const float* inp  = (const float*)d_in[0];
  const float* mask = (const float*)d_in[1];
  const int*   row  = (const int*)d_in[2];
  const int*   col  = (const int*)d_in[3];
  const float* hW   = (const float*)d_in[4];
  const float* hb   = (const float*)d_in[5];
  const float* gW   = (const float*)d_in[6];
  const float* gb   = (const float*)d_in[7];
  const float* qW   = (const float*)d_in[8];
  const float* qb   = (const float*)d_in[9];
  const float* kW   = (const float*)d_in[10];
  const float* kb   = (const float*)d_in[11];
  const float* s1W  = (const float*)d_in[12];
  const float* s1b  = (const float*)d_in[13];
  const float* s2W  = (const float*)d_in[14];
  const float* s2b  = (const float*)d_in[15];
  float* out = (float*)d_out;

  char* ws = (char*)d_ws;
  bf16_t* hWt  = (bf16_t*)ws; ws += 32768;
  bf16_t* qWt  = (bf16_t*)ws; ws += 32768;
  bf16_t* gWt  = (bf16_t*)ws; ws += 32768;
  bf16_t* s1Wt = (bf16_t*)ws; ws += 32768;
  bf16_t* inpT = (bf16_t*)ws; ws += (size_t)BATCH * 2048 * 64 * 2;
  bf16_t* kbf  = (bf16_t*)ws; ws += (size_t)BATCH * RR * 2048 * 2;

  prep_wt  <<<256, 256, 0, stream>>>(hW, qW, gW, s1W, hWt, qWt, gWt, s1Wt);
  prep_inpT<<<4096, 256, 0, stream>>>(inp, inpT);
  prep_k   <<<4096, 256, 0, stream>>>(inp, kW, kb, kbf);
  phylo_main<<<BATCH * NTILES, 256, 0, stream>>>(
      inp, mask, row, col, hb, gb, qb, s1b, s2W, s2b,
      hWt, qWt, gWt, s1Wt, kbf, inpT, out);
}
